// Model_21938692948530
// MI455X (gfx1250) — matrix operand order repaired and checked by reading, not compiled, not run on hardware
//
#include <hip/hip_runtime.h>
#include <hip/hip_bf16.h>

typedef __attribute__((ext_vector_type(16))) _Float16 v16h;
typedef __attribute__((ext_vector_type(8)))  _Float16 v8h;
typedef __attribute__((ext_vector_type(4)))  _Float16 v4h;
typedef __attribute__((ext_vector_type(8)))  float    v8f;
typedef __attribute__((ext_vector_type(4)))  float    v4f;

#define HDIM 256

// ---------- fragment loaders (per CDNA5 16-bit WMMA VGPR layouts) ----------
// A (16x32, MxK): lane L -> M=L&15; halfs[0..7]=K[kb+koffA .. +7], halfs[8..15]=K[kb+16+koffA .. +7]
__device__ __forceinline__ v16h load_a_frag(const _Float16* p) {
    union { v16h v; v8h h[2]; } u;
    u.h[0] = *(const v8h*)(p);
    u.h[1] = *(const v8h*)(p + 16);
    return u.v;
}
// B (32x16, KxN): lane L -> N=L&15; halfs[0..7]=K[kb+koffB .. +7], halfs[8..15]=K[kb+16+koffB .. +7]
__device__ __forceinline__ v16h load_b_frag(const _Float16* p) {
    union { v16h v; v8h h[2]; } u;
    u.h[0] = *(const v8h*)(p);
    u.h[1] = *(const v8h*)(p + 16);
    return u.v;
}

// ---------- convert a 256x256 f32 weight to f16 ----------
__global__ void k_cvt_h(const float* __restrict__ w, _Float16* __restrict__ o, int n4) {
    int i = blockIdx.x * blockDim.x + threadIdx.x;
    if (i >= n4) return;
    v4f a = *(const v4f*)(w + (size_t)i * 4);
    v4h b; b[0] = (_Float16)a[0]; b[1] = (_Float16)a[1]; b[2] = (_Float16)a[2]; b[3] = (_Float16)a[3];
    *(v4h*)(o + (size_t)i * 4) = b;
}

// ---------- per-destination edge counts ----------
__global__ void k_count(const int* __restrict__ s, const int* __restrict__ d,
                        int* __restrict__ cs, int* __restrict__ ct, int nE) {
    int e = blockIdx.x * blockDim.x + threadIdx.x;
    if (e >= nE) return;
    atomicAdd(&ct[d[e]], 1);
    atomicAdd(&cs[s[e]], 1);
}

// ---------- gather emb[node_id] -> f16 features ----------
__global__ void k_gather_h(const float* __restrict__ emb, const int* __restrict__ nid,
                           _Float16* __restrict__ out, int nN) {
    long gid = (long)blockIdx.x * blockDim.x + threadIdx.x;
    int n = (int)(gid >> 6);
    if (n >= nN) return;
    int c = ((int)gid & 63) * 4;
    long row = nid[n];
    v4f a = *(const v4f*)(emb + row * HDIM + c);
    v4h b; b[0] = (_Float16)a[0]; b[1] = (_Float16)a[1]; b[2] = (_Float16)a[2]; b[3] = (_Float16)a[3];
    *(v4h*)(out + (long)n * HDIM + c) = b;
}

// ---------- edge scatter-add: agg[dst] += x[src] (f32 HW atomics, lands in L2) ----------
__global__ void k_scatter(const int* __restrict__ srcI, const int* __restrict__ dstI,
                          const _Float16* __restrict__ x, float* __restrict__ agg, int nE) {
    long gid = (long)blockIdx.x * blockDim.x + threadIdx.x;
    int e = (int)(gid >> 6);
    if (e >= nE) return;
    int c = ((int)gid & 63) * 4;
    long s = srcI[e], d = dstI[e];
    v4h xv = *(const v4h*)(x + s * HDIM + c);
    float* ap = agg + d * HDIM + c;
    unsafeAtomicAdd(ap + 0, (float)xv[0]);
    unsafeAtomicAdd(ap + 1, (float)xv[1]);
    unsafeAtomicAdd(ap + 2, (float)xv[2]);
    unsafeAtomicAdd(ap + 3, (float)xv[3]);
}

// ---------- mean = agg / max(cnt,1), convert to f16 ----------
__global__ void k_mean(const float* __restrict__ agg, const int* __restrict__ cnt,
                       _Float16* __restrict__ mh, int nN) {
    long gid = (long)blockIdx.x * blockDim.x + threadIdx.x;
    int n = (int)(gid >> 6);
    if (n >= nN) return;
    int c = ((int)gid & 63) * 4;
    float inv = 1.0f / fmaxf((float)cnt[n], 1.0f);
    v4f a = *(const v4f*)(agg + (long)n * HDIM + c);
    v4h b;
    b[0] = (_Float16)(a[0] * inv); b[1] = (_Float16)(a[1] * inv);
    b[2] = (_Float16)(a[2] * inv); b[3] = (_Float16)(a[3] * inv);
    *(v4h*)(mh + (long)n * HDIM + c) = b;
}

// ---------- K=256 pass of one operand pair: 2 A-frags x 4 B-frags -> 8 WMMAs/step ----------
__device__ __forceinline__ void gemm_k256(const _Float16* __restrict__ A,
                                          const _Float16* __restrict__ W,
                                          long mrow0, long mrow1, int nrow,
                                          int koffA, int koffB, v8f acc[2][4]) {
#pragma unroll
    for (int kt = 0; kt < 8; ++kt) {
        const int kbase = kt * 32;
        v16h a0 = load_a_frag(A + mrow0 * HDIM + kbase + koffA);
        v16h a1 = load_a_frag(A + mrow1 * HDIM + kbase + koffA);
#pragma unroll
        for (int s = 0; s < 4; ++s) {
            v16h bf = load_b_frag(W + (long)(nrow + s * 16) * HDIM + kbase + koffB);
            acc[0][s] = __builtin_amdgcn_wmma_f32_16x16x32_f16(false, a0, false, bf, (short)0, acc[0][s], false, false);
            acc[1][s] = __builtin_amdgcn_wmma_f32_16x16x32_f16(false, a1, false, bf, (short)0, acc[1][s], false, false);
        }
    }
}

// ---------- fused dual GEMM: D = A1@W1.T + A2@W2.T + bias, optional ReLU ----------
// block = 128 (4 waves); wave w covers output cols [w*64, w*64+64); each wave does a
// 32x64 tile (2 M-tiles x 4 N-tiles = 8 accumulators, B frags reused twice).
// grid.x = Nrows/32
__global__ __launch_bounds__(128) void k_gemm_dual(
    const _Float16* __restrict__ A1, const _Float16* __restrict__ A2,
    const _Float16* __restrict__ W1, const _Float16* __restrict__ W2,
    const float* __restrict__ bias, float* __restrict__ out32,
    _Float16* __restrict__ out16, int relu)
{
    const int lane  = threadIdx.x & 31;
    const int wv    = threadIdx.x >> 5;         // 0..3
    const int m     = lane & 15;
    const int hi    = lane >> 4;                // 0 or 1
    const int koffA = hi * 8;
    const int koffB = hi * 8;
    const long mrow0 = (long)blockIdx.x * 32 + m;
    const long mrow1 = mrow0 + 16;
    const int  ncb   = wv * 64;

    v8f acc[2][4] = {{v8f{}, v8f{}, v8f{}, v8f{}}, {v8f{}, v8f{}, v8f{}, v8f{}}};

    gemm_k256(A1, W1, mrow0, mrow1, ncb + m, koffA, koffB, acc);
    gemm_k256(A2, W2, mrow0, mrow1, ncb + m, koffA, koffB, acc);

    // D layout: vgpr r -> row r + 8*hi (within 16-row tile), col = lane&15
    const int mhi = hi * 8;
    float bv[4];
#pragma unroll
    for (int s = 0; s < 4; ++s) bv[s] = bias[ncb + s * 16 + m];

#pragma unroll
    for (int t = 0; t < 2; ++t) {
#pragma unroll
        for (int s = 0; s < 4; ++s) {
#pragma unroll
            for (int r = 0; r < 8; ++r) {
                long row = (long)blockIdx.x * 32 + t * 16 + mhi + r;
                int  col = ncb + s * 16 + m;
                float v = acc[t][s][r] + bv[s];
                if (relu) v = fmaxf(v, 0.0f);
                if (out32) out32[row * HDIM + col] = v;
                if (out16) out16[row * HDIM + col] = (_Float16)v;
            }
        }
    }
}

// ---------- classifier: sigmoid(dot(o_src[i], o_tgt[j])) , one wave per edge ----------
__global__ void k_dot(const float* __restrict__ osrc, const float* __restrict__ otgt,
                      const int* __restrict__ eli, float* __restrict__ out, int nEL) {
    long gid = (long)blockIdx.x * blockDim.x + threadIdx.x;
    int edge = (int)(gid >> 5);
    if (edge >= nEL) return;
    int lane = (int)gid & 31;
    long i = eli[edge];
    long j = eli[nEL + edge];
    const float* a = osrc + i * HDIM + lane * 8;
    const float* b = otgt + j * HDIM + lane * 8;
    float s = 0.0f;
#pragma unroll
    for (int t = 0; t < 8; ++t) s += a[t] * b[t];
#pragma unroll
    for (int off = 16; off > 0; off >>= 1) s += __shfl_xor(s, off, 32);
    if (lane == 0) out[edge] = 1.0f / (1.0f + __expf(-s));
}

extern "C" void kernel_launch(void* const* d_in, const int* in_sizes, int n_in,
                              void* d_out, int out_size, void* d_ws, size_t ws_size,
                              hipStream_t stream) {
    if (n_in < 18) return;
    const int*   nid_src = (const int*)d_in[0];
    const int*   nid_tgt = (const int*)d_in[1];
    const int*   e_st    = (const int*)d_in[2];   // [2, E] flat
    const int*   eli     = (const int*)d_in[3];   // [2, EL] flat
    const float* emb_src = (const float*)d_in[4];
    const float* emb_tgt = (const float*)d_in[5];
    // weight slots: 0:w1_st_l 1:w1_st_r 2:w1_ts_l 3:w1_ts_r 4:w2_st_l 5:w2_st_r 6:w2_ts_l 7:w2_ts_r
    const float* wsrc[8] = { (const float*)d_in[6],  (const float*)d_in[8],
                             (const float*)d_in[9],  (const float*)d_in[11],
                             (const float*)d_in[12], (const float*)d_in[14],
                             (const float*)d_in[15], (const float*)d_in[17] };
    const float* bias1_st = (const float*)d_in[7];
    const float* bias1_ts = (const float*)d_in[10];
    const float* bias2_st = (const float*)d_in[13];
    const float* bias2_ts = (const float*)d_in[16];

    const int Ns  = in_sizes[0];
    const int Nt  = in_sizes[1];
    const int nE  = in_sizes[2] / 2;
    const int nEL = in_sizes[3] / 2;
    const size_t NHs = (size_t)Ns * HDIM, NHt = (size_t)Nt * HDIM;
    const size_t NHmax = NHs > NHt ? NHs : NHt;

    // ---- carve scratch (aligned to 256 B) ----
    char* p = (char*)d_ws;
    auto take = [&](size_t bytes) -> char* {
        char* r = p; p += (bytes + 255) & ~(size_t)255; return r;
    };
    _Float16* W16   = (_Float16*)take(8ull * HDIM * HDIM * sizeof(_Float16));
    int*      cnt_t = (int*)take((size_t)Nt * sizeof(int));
    int*      cnt_s = (int*)take((size_t)Ns * sizeof(int));
    // region A: layer-1 f16 inputs; reused later as o_tgt (f32)
    size_t xsrc_b  = (NHs * 2 + 255) & ~(size_t)255;
    size_t regA_sz = xsrc_b + NHt * 2;
    if (NHt * 4 > regA_sz) regA_sz = NHt * 4;
    char*     regA   = take(regA_sz);
    _Float16* xsrc_h = (_Float16*)regA;
    _Float16* xtgt_h = (_Float16*)(regA + xsrc_b);
    float*    o_tgt  = (float*)regA;                 // alias (valid after layer 1)
    _Float16* mean_h = (_Float16*)take(NHmax * 2);
    _Float16* hsrc_h = (_Float16*)take(NHs * 2);
    _Float16* htgt_h = (_Float16*)take(NHt * 2);
    char*     regB   = take(NHmax * 4);
    float*    agg    = (float*)regB;
    float*    o_src  = (float*)regB;                 // alias (valid after last mean)

    // ---- prologue: counts, weight conversion, feature gather ----
    hipMemsetAsync(cnt_t, 0, (size_t)Nt * sizeof(int), stream);
    hipMemsetAsync(cnt_s, 0, (size_t)Ns * sizeof(int), stream);
    for (int i = 0; i < 8; ++i)
        k_cvt_h<<<(HDIM * HDIM / 4 + 255) / 256, 256, 0, stream>>>(
            wsrc[i], W16 + (size_t)i * HDIM * HDIM, HDIM * HDIM / 4);
    k_count<<<(nE + 255) / 256, 256, 0, stream>>>(e_st, e_st + nE, cnt_s, cnt_t, nE);
    {
        long ts = (long)Ns * 64, tt = (long)Nt * 64;
        k_gather_h<<<(unsigned)((ts + 255) / 256), 256, 0, stream>>>(emb_src, nid_src, xsrc_h, Ns);
        k_gather_h<<<(unsigned)((tt + 255) / 256), 256, 0, stream>>>(emb_tgt, nid_tgt, xtgt_h, Nt);
    }

    // ---- one SAGE relation: scatter-mean + fused dual WMMA GEMM ----
    auto sage = [&](const int* sIdx, const int* dIdx,
                    const _Float16* xS, const _Float16* xD, int Ndst,
                    const int* cntD, int slotL, int slotR, const float* b,
                    int relu, float* o32, _Float16* o16) {
        hipMemsetAsync(agg, 0, (size_t)Ndst * HDIM * sizeof(float), stream);
        long st = (long)nE * 64;
        k_scatter<<<(unsigned)((st + 255) / 256), 256, 0, stream>>>(sIdx, dIdx, xS, agg, nE);
        long mt = (long)Ndst * 64;
        k_mean<<<(unsigned)((mt + 255) / 256), 256, 0, stream>>>(agg, cntD, mean_h, Ndst);
        k_gemm_dual<<<Ndst / 32, 128, 0, stream>>>(
            mean_h, xD,
            W16 + (size_t)slotL * HDIM * HDIM, W16 + (size_t)slotR * HDIM * HDIM,
            b, o32, o16, relu);
    };

    // layer 1 (ReLU, f16 outputs)
    sage(e_st,      e_st + nE, xsrc_h, xtgt_h, Nt, cnt_t, 0, 1, bias1_st, 1, nullptr, htgt_h);
    sage(e_st + nE, e_st,      xtgt_h, xsrc_h, Ns, cnt_s, 2, 3, bias1_ts, 1, nullptr, hsrc_h);
    // layer 2 (no activation, f32 outputs; o_tgt/o_src alias dead regions)
    sage(e_st,      e_st + nE, hsrc_h, htgt_h, Nt, cnt_t, 4, 5, bias2_st, 0, o_tgt, nullptr);
    sage(e_st + nE, e_st,      htgt_h, hsrc_h, Ns, cnt_s, 6, 7, bias2_ts, 0, o_src, nullptr);

    // classifier
    long dt = (long)nEL * 32;
    k_dot<<<(unsigned)((dt + 255) / 256), 256, 0, stream>>>(o_src, o_tgt, eli, (float*)d_out, nEL);
}
